// ConceptTransformer_62835371540499
// MI455X (gfx1250) — compile-verified, matrix operand order checked by reading, not run on hardware
//
#include <hip/hip_runtime.h>
#include <hip/hip_bf16.h>
#include <math.h>

// ---------------------------------------------------------------------------
// Types for CDNA5 WMMA (wave32, 16x16x32 bf16 -> f32)
// ---------------------------------------------------------------------------
typedef __bf16 bf16;
typedef __attribute__((ext_vector_type(8)))  __bf16 bf16x8;
typedef __attribute__((ext_vector_type(16))) __bf16 v16bf;
typedef __attribute__((ext_vector_type(8)))  float  v8f;
typedef __attribute__((ext_vector_type(4)))  float  f32x4;

#define V8F_ZERO {0.f, 0.f, 0.f, 0.f, 0.f, 0.f, 0.f, 0.f}

__device__ __forceinline__ v8f wmma_bf16(v16bf a, v16bf b, v8f c) {
    return __builtin_amdgcn_wmma_f32_16x16x32_bf16(false, a, false, b, (short)0, c,
                                                   false, false);
}

// Load a 16x32 bf16 fragment (A-operand layout; also B-operand when source is
// B^T row-major).  ISA layout: lane&15 = row(M/N), half-wave selects K 0-7/16-23
// vs 8-15/24-31, so each lane does two contiguous 16B loads.
__device__ __forceinline__ v16bf frag_ld(const bf16* base, int ld) {
    const int lane = threadIdx.x & 31;
    const int row  = lane & 15;
    const int kb   = (lane >> 4) << 3;      // 0 or 8
    const bf16* p  = base + row * ld + kb;
    bf16x8 lo = *(const bf16x8*)(p);
    bf16x8 hi = *(const bf16x8*)(p + 16);
    v16bf r;
#pragma unroll
    for (int e = 0; e < 8; ++e) { r[e] = lo[e]; r[e + 8] = hi[e]; }
    return r;
}

// D-fragment element mapping: slot e of lane l -> M = e + 8*(l>>4), N = l&15.

__device__ __forceinline__ int iclip7(int v) { return v < -7 ? -7 : (v > 7 ? 7 : v); }

// ---------------------------------------------------------------------------
// Problem constants
// ---------------------------------------------------------------------------
#define BB  8
#define NN  1024
#define CC  256
#define HH  8
#define HD  32
#define LP  256      // padded bucket count (real L = 225)
#define BHN 64       // B*H

// ---------------------------------------------------------------------------
// Small conversion / init kernels
// ---------------------------------------------------------------------------
__global__ void k_f2bf(bf16* __restrict__ d, const float* __restrict__ s, int n) {
    int i = blockIdx.x * blockDim.x + threadIdx.x;
    if (i < n) d[i] = (bf16)s[i];
}
__global__ void k_zero(float* __restrict__ p, int n) {
    int i = blockIdx.x * blockDim.x + threadIdx.x;
    if (i < n) p[i] = 0.f;
}
// src [8][225][32] f32 -> dst [8][256][32] bf16 (zero padded)
__global__ void k_tab_pad(bf16* __restrict__ d, const float* __restrict__ s) {
    int i = blockIdx.x * blockDim.x + threadIdx.x;
    if (i >= HH * LP * HD) return;
    int h = i >> 13, l = (i >> 5) & 255, dd = i & 31;
    d[i] = (bf16)(l < 225 ? s[((size_t)h * 225 + l) * 32 + dd] : 0.f);
}
// src [8][225][32] f32 -> dst [8][32][256] bf16 (transposed, zero padded)
__global__ void k_tab_pad_t(bf16* __restrict__ d, const float* __restrict__ s) {
    int i = blockIdx.x * blockDim.x + threadIdx.x;
    if (i >= HH * HD * LP) return;
    int h = i >> 13, dd = (i >> 8) & 31, l = i & 255;
    d[i] = (bf16)(l < 225 ? s[((size_t)h * 225 + l) * 32 + dd] : 0.f);
}

// ---------------------------------------------------------------------------
// QKV GEMM: [8192,256]bf16 @ [768,256]^T.  One wave = 16x64 tile strip:
// the A fragment is loaded once per K-step and reused by 4 WMMAs.
// Results scatter into q/k (bh,n,d) and v transposed (bh,d,n).
// ---------------------------------------------------------------------------
__global__ __launch_bounds__(32) void k_gemm_qkv(const bf16* __restrict__ A,
                                                 const bf16* __restrict__ W,
                                                 bf16* __restrict__ q,
                                                 bf16* __restrict__ k,
                                                 bf16* __restrict__ vt) {
    const int blk = blockIdx.x;            // 12 * 512
    const int tg = blk % 12, ty = blk / 12;
    v8f acc[4] = {V8F_ZERO, V8F_ZERO, V8F_ZERO, V8F_ZERO};
#pragma unroll
    for (int ks = 0; ks < 8; ++ks) {
        v16bf af = frag_ld(A + (size_t)ty * 16 * 256 + ks * 32, 256);
#pragma unroll
        for (int t = 0; t < 4; ++t) {
            v16bf bw = frag_ld(W + (size_t)(tg * 4 + t) * 16 * 256 + ks * 32, 256);
            acc[t] = wmma_bf16(af, bw, acc[t]);
        }
    }
    const int lane = threadIdx.x, nn = lane & 15, half = lane >> 4;
#pragma unroll
    for (int t = 0; t < 4; ++t) {
        const int o = (tg * 4 + t) * 16 + nn;
        const int mat = o >> 8, c = o & 255, hh = c >> 5, dd = c & 31;
#pragma unroll
        for (int e = 0; e < 8; ++e) {
            int m = ty * 16 + e + 8 * half;
            int b = m >> 10, n = m & 1023;
            int bh = b * HH + hh;
            bf16 val = (bf16)acc[t][e];
            if (mat == 0)      q[(size_t)(bh * NN + n) * HD + dd] = val;
            else if (mat == 1) k[(size_t)(bh * NN + n) * HD + dd] = val;
            else               vt[((size_t)bh * HD + dd) * NN + n] = val;
        }
    }
}

// ---------------------------------------------------------------------------
// Bucket-projection GEMM: out[bh][i][l] = scl * (X[bh][i][:] . tab[h][l][:])
// K = HD = 32.  One wave = 16x64 strip (A fragment reused by 4 WMMAs).
// ---------------------------------------------------------------------------
__global__ __launch_bounds__(32) void k_gemm_pos(const bf16* __restrict__ X,
                                                 const bf16* __restrict__ tab,
                                                 bf16* __restrict__ out, float scl) {
    const int blk = blockIdx.x;            // bh(64) * it(64) * lg(4)
    const int lg = blk & 3, it = (blk >> 2) & 63, bh = blk >> 8;
    const int h = bh & 7;
    v16bf af = frag_ld(X + (size_t)(bh * NN + it * 16) * HD, HD);
    v8f acc[4];
#pragma unroll
    for (int t = 0; t < 4; ++t) {
        v16bf bt = frag_ld(tab + ((size_t)h * LP + (lg * 4 + t) * 16) * HD, HD);
        v8f z = V8F_ZERO;
        acc[t] = wmma_bf16(af, bt, z);
    }
    const int lane = threadIdx.x, nn = lane & 15, half = lane >> 4;
#pragma unroll
    for (int t = 0; t < 4; ++t) {
        const int l = (lg * 4 + t) * 16 + nn;
#pragma unroll
        for (int e = 0; e < 8; ++e) {
            int i = it * 16 + e + 8 * half;
            out[(size_t)(bh * NN + i) * LP + l] = (bf16)(acc[t][e] * scl);
        }
    }
}

// ---------------------------------------------------------------------------
// Fused iRPE flash attention.  One wave32 per (bh, 16-row block).
//
// Key invariants exploited: a 16-row i-block never crosses a 32-wide grid row
// (ri is constant per block) and a 32-j step is exactly one grid row (rj is a
// scalar per step, cj = nn + 16t).  Bucket ids reduce to a1 + clip(ci-cj) /
// a2 - clip(ci-cj) with scalar a1,a2 -> one v_med3 per element.
//
// Pass 1 keeps lane-local online (m,s) per D-slot (no LDS, no barriers); the
// 16-lane shfl_xor reduction happens once.  Pass 2 recomputes logits, emits
// normalized probs straight from fragment layout (P tile + ds_add_f32 bucket
// scatter), runs PV WMMAs, then folds in A @ rpe_v^T with 16 more WMMAs.
// ---------------------------------------------------------------------------
__global__ __launch_bounds__(32) void k_attn(const bf16* __restrict__ qb,
                                             const bf16* __restrict__ kb,
                                             const bf16* __restrict__ vt,
                                             const bf16* __restrict__ tk,
                                             const bf16* __restrict__ tq,
                                             const bf16* __restrict__ rvt,
                                             bf16* __restrict__ ctx) {
    __shared__ bf16  Pt[16 * 40];        // probability tile (A-frag source)
    __shared__ bf16  tkL[16 * LP];       // tk rows for this i-block (8 KB)
    __shared__ float Abuf[16 * LP];      // bucket accumulator (16 KB)
    __shared__ __attribute__((aligned(16))) float mrow[16];
    __shared__ __attribute__((aligned(16))) float invs[16];

    const int lane = threadIdx.x;
    const int blk  = blockIdx.x;
    const int ib   = blk & 63;
    const int bh   = blk >> 6;
    const int h    = bh & 7;
    const int i0   = ib << 4;
    const int nn   = lane & 15, half = lane >> 4;

    {   // preload tk rows (16 x 256 bf16 = 8 KB) and zero Abuf
        const f32x4* s4 = (const f32x4*)(tk + ((size_t)(bh * NN + i0)) * LP);
        f32x4* d4 = (f32x4*)tkL;
        for (int t = lane; t < (16 * LP * 2) / 16; t += 32) d4[t] = s4[t];
        for (int t = lane; t < 16 * LP; t += 32) Abuf[t] = 0.f;
    }
    __syncthreads();

    // Persistent q fragment (whole K dim = HD = 32 fits one fragment)
    const v16bf aq = frag_ld(qb + (size_t)(bh * NN + i0) * HD, HD);

    const int ri0 = i0 >> 5;                    // constant grid row of i-block
    const int cib = (i0 & 31) + 8 * half;       // ci = cib + e

    // Computes the biased 16x32 logit tile for this j-step; also returns the
    // tk-side bucket ids (needed for the pass-2 scatter).
    auto step_S = [&](int j0, v8f& S0, v8f& S1, int bij0[8], int bij1[8]) {
        v16bf kf0 = frag_ld(kb + (size_t)(bh * NN + j0) * HD, HD);
        v16bf kf1 = frag_ld(kb + (size_t)(bh * NN + j0 + 16) * HD, HD);
        v8f z = V8F_ZERO;
        S0 = wmma_bf16(aq, kf0, z);
        S1 = wmma_bf16(aq, kf1, z);
        const int rj = j0 >> 5;                 // scalar per step
        const int dr = iclip7(ri0 - rj);
        const int a1 = (dr + 7) * 15 + 7;       // bucket(i,j) = a1 + dc
        const int a2 = (7 - dr) * 15 + 7;       // bucket(j,i) = a2 - dc
        const bf16* tq0 = tq + (size_t)(bh * NN + j0 + nn) * LP + a2;
        const bf16* tq1 = tq0 + (size_t)16 * LP;
        const int d0 = cib - nn;                // dc0 = clip(d0 + e)
        const int d1 = d0 - 16;                 // dc1 = clip(d1 + e)
#pragma unroll
        for (int e = 0; e < 8; ++e) {
            const int dc0 = iclip7(d0 + e);
            const int dc1 = iclip7(d1 + e);
            bij0[e] = a1 + dc0;
            bij1[e] = a1 + dc1;
            const bf16* tkr = &tkL[(e + 8 * half) * LP];
            S0[e] += (float)tkr[bij0[e]] + (float)tq0[-dc0];
            S1[e] += (float)tkr[bij1[e]] + (float)tq1[-dc1];
        }
    };

    // ---- Pass 1: lane-local online max/sum, barrier-free ----
    float mL[8], sL[8];
#pragma unroll
    for (int e = 0; e < 8; ++e) { mL[e] = -1e30f; sL[e] = 0.f; }
    for (int j0 = 0; j0 < NN; j0 += 32) {
        v8f S0, S1; int b0[8], b1[8];
        step_S(j0, S0, S1, b0, b1);
#pragma unroll
        for (int e = 0; e < 8; ++e) {
            float t  = fmaxf(S0[e], S1[e]);
            float mn = fmaxf(mL[e], t);
            sL[e] = sL[e] * __expf(mL[e] - mn) + __expf(S0[e] - mn) + __expf(S1[e] - mn);
            mL[e] = mn;
        }
    }
    // one-shot 16-lane reduction within each half-wave (rows 8*half + e)
#pragma unroll
    for (int e = 0; e < 8; ++e) {
        float mf = mL[e];
        mf = fmaxf(mf, __shfl_xor(mf, 1, 32));
        mf = fmaxf(mf, __shfl_xor(mf, 2, 32));
        mf = fmaxf(mf, __shfl_xor(mf, 4, 32));
        mf = fmaxf(mf, __shfl_xor(mf, 8, 32));
        float sf = sL[e] * __expf(mL[e] - mf);
        sf += __shfl_xor(sf, 1, 32);
        sf += __shfl_xor(sf, 2, 32);
        sf += __shfl_xor(sf, 4, 32);
        sf += __shfl_xor(sf, 8, 32);
        if (nn == 0) { mrow[8 * half + e] = mf; invs[8 * half + e] = 1.f / sf; }
    }
    __syncthreads();
    float mF[8], iF[8];
#pragma unroll
    for (int e = 0; e < 8; ++e) { mF[e] = mrow[8 * half + e]; iF[e] = invs[8 * half + e]; }

    // ---- Pass 2: final probs, bucket scatter, PV WMMAs ----
    v8f acc0 = V8F_ZERO;
    v8f acc1 = V8F_ZERO;
    for (int j0 = 0; j0 < NN; j0 += 32) {
        v8f S0, S1; int b0[8], b1[8];
        step_S(j0, S0, S1, b0, b1);
#pragma unroll
        for (int e = 0; e < 8; ++e) {
            const int row = e + 8 * half;
            float p0 = __expf(S0[e] - mF[e]) * iF[e];
            float p1 = __expf(S1[e] - mF[e]) * iF[e];
            Pt[row * 40 + nn]      = (bf16)p0;
            Pt[row * 40 + 16 + nn] = (bf16)p1;
            atomicAdd(&Abuf[row * LP + b0[e]], p0);   // ds_add_f32
            atomicAdd(&Abuf[row * LP + b1[e]], p1);
        }
        __syncthreads();
        v16bf pa  = frag_ld(Pt, 40);
        v16bf vf0 = frag_ld(vt + ((size_t)bh * HD + 0)  * NN + j0, NN);
        v16bf vf1 = frag_ld(vt + ((size_t)bh * HD + 16) * NN + j0, NN);
        acc0 = wmma_bf16(pa, vf0, acc0);
        acc1 = wmma_bf16(pa, vf1, acc1);
        __syncthreads();
    }

    // ---- out += Abuf @ rpe_v^T  (K = 256, 8 WMMA steps) ----
#pragma unroll
    for (int ks = 0; ks < 8; ++ks) {
        const float* ap = &Abuf[nn * LP + (half << 3) + ks * 32];
        v16bf af;
#pragma unroll
        for (int e = 0; e < 8; ++e) { af[e] = (bf16)ap[e]; af[e + 8] = (bf16)ap[16 + e]; }
        v16bf b0 = frag_ld(rvt + ((size_t)h * HD + 0)  * LP + ks * 32, LP);
        v16bf b1 = frag_ld(rvt + ((size_t)h * HD + 16) * LP + ks * 32, LP);
        acc0 = wmma_bf16(af, b0, acc0);
        acc1 = wmma_bf16(af, b1, acc1);
    }

    // ---- store context in [b][n][c] bf16 layout for the proj GEMM ----
    const int b = bh >> 3;
#pragma unroll
    for (int e = 0; e < 8; ++e) {
        const int i = i0 + e + 8 * half;
        const size_t base = ((size_t)(b * NN + i)) * CC + h * HD;
        ctx[base + nn]      = (bf16)acc0[e];
        ctx[base + 16 + nn] = (bf16)acc1[e];
    }
}

// ---------------------------------------------------------------------------
// Proj GEMM + bias + residual + mean-pool.  One wave = 16x64 strip; the
// 16-row tile is reduced in registers (shfl_xor 16) before one atomic per col.
// ---------------------------------------------------------------------------
__global__ __launch_bounds__(32) void k_proj(const bf16* __restrict__ A,
                                             const bf16* __restrict__ W,
                                             const float* __restrict__ pb,
                                             const float* __restrict__ x,
                                             float* __restrict__ pooled) {
    const int blk = blockIdx.x;            // 4 * 512
    const int tg = blk & 3, ty = blk >> 2;
    v8f acc[4] = {V8F_ZERO, V8F_ZERO, V8F_ZERO, V8F_ZERO};
#pragma unroll
    for (int ks = 0; ks < 8; ++ks) {
        v16bf af = frag_ld(A + (size_t)ty * 16 * 256 + ks * 32, 256);
#pragma unroll
        for (int t = 0; t < 4; ++t) {
            v16bf bw = frag_ld(W + (size_t)(tg * 4 + t) * 16 * 256 + ks * 32, 256);
            acc[t] = wmma_bf16(af, bw, acc[t]);
        }
    }
    const int lane = threadIdx.x, nn = lane & 15, half = lane >> 4;
    const int b = (ty * 16) >> 10;         // whole tile lies in one batch image
#pragma unroll
    for (int t = 0; t < 4; ++t) {
        const int o = (tg * 4 + t) * 16 + nn;
        const float bias = pb[o];
        float part = 0.f;
#pragma unroll
        for (int e = 0; e < 8; ++e) {
            const int m = ty * 16 + e + 8 * half;
            part += acc[t][e] + bias + x[(size_t)m * CC + o];
        }
        part += __shfl_xor(part, 16, 32);
        if (half == 0) atomicAdd(&pooled[b * CC + o], part * (1.0f / (float)NN));
    }
}

// ---------------------------------------------------------------------------
// Classification head: out[b,cls] = pooled[b,:] . fc_w[cls,:] + fc_b[cls]
// ---------------------------------------------------------------------------
__global__ void k_head(const float* __restrict__ pooled, const float* __restrict__ fcw,
                       const float* __restrict__ fcb, float* __restrict__ out) {
    const int t = blockIdx.x * blockDim.x + threadIdx.x;
    if (t >= BB * 15) return;
    const int b = t / 15, c = t % 15;
    float s = fcb[c];
    for (int e = 0; e < CC; ++e) s += pooled[b * CC + e] * fcw[c * CC + e];
    out[b * 15 + c] = s;
}

// ---------------------------------------------------------------------------
// Host-side launch
// ---------------------------------------------------------------------------
extern "C" void kernel_launch(void* const* d_in, const int* in_sizes, int n_in,
                              void* d_out, int out_size, void* d_ws, size_t ws_size,
                              hipStream_t stream) {
    (void)in_sizes; (void)n_in; (void)out_size; (void)ws_size;
    const float* x      = (const float*)d_in[0];
    const float* qkv_w  = (const float*)d_in[1];
    const float* proj_w = (const float*)d_in[2];
    const float* proj_b = (const float*)d_in[3];
    const float* fc_w   = (const float*)d_in[4];
    const float* fc_b   = (const float*)d_in[5];
    const float* rpe_k  = (const float*)d_in[6];
    const float* rpe_q  = (const float*)d_in[7];
    const float* rpe_v  = (const float*)d_in[8];
    // d_in[9] = rp_bucket is recomputed analytically on-device.
    float* out = (float*)d_out;

    char* ws = (char*)d_ws;
    size_t off = 0;
    auto carve = [&](size_t bytes) { void* p = ws + off; off = (off + bytes + 255) & ~size_t(255); return p; };
    bf16* xbf    = (bf16*)carve((size_t)BB * NN * CC * 2);        // 4 MiB
    bf16* qkvwbf = (bf16*)carve((size_t)3 * CC * CC * 2);
    bf16* projwbf= (bf16*)carve((size_t)CC * CC * 2);
    bf16* rpekbf = (bf16*)carve((size_t)HH * LP * HD * 2);
    bf16* rpeqbf = (bf16*)carve((size_t)HH * LP * HD * 2);
    bf16* rvt    = (bf16*)carve((size_t)HH * HD * LP * 2);
    bf16* qB     = (bf16*)carve((size_t)BHN * NN * HD * 2);       // 4 MiB
    bf16* kB     = (bf16*)carve((size_t)BHN * NN * HD * 2);
    bf16* vT     = (bf16*)carve((size_t)BHN * HD * NN * 2);
    bf16* tk     = (bf16*)carve((size_t)BHN * NN * LP * 2);       // 32 MiB
    bf16* tq     = (bf16*)carve((size_t)BHN * NN * LP * 2);       // 32 MiB
    bf16* ctx    = (bf16*)carve((size_t)BB * NN * CC * 2);
    float* pooled= (float*)carve((size_t)BB * CC * 4);

    const float scale = 0.17677669529663689f;  // HD^-0.5

    // conversions / padding / init
    k_f2bf   <<<(BB*NN*CC + 255) / 256, 256, 0, stream>>>(xbf, x, BB*NN*CC);
    k_f2bf   <<<(3*CC*CC  + 255) / 256, 256, 0, stream>>>(qkvwbf, qkv_w, 3*CC*CC);
    k_f2bf   <<<(CC*CC    + 255) / 256, 256, 0, stream>>>(projwbf, proj_w, CC*CC);
    k_tab_pad<<<(HH*LP*HD + 255) / 256, 256, 0, stream>>>(rpekbf, rpe_k);
    k_tab_pad<<<(HH*LP*HD + 255) / 256, 256, 0, stream>>>(rpeqbf, rpe_q);
    k_tab_pad_t<<<(HH*HD*LP + 255) / 256, 256, 0, stream>>>(rvt, rpe_v);
    k_zero   <<<(BB*CC + 255) / 256, 256, 0, stream>>>(pooled, BB*CC);

    // QKV projection (16x64 strip per wave)
    k_gemm_qkv<<<12 * 512, 32, 0, stream>>>(xbf, qkvwbf, qB, kB, vT);

    // bucket projections: tk = q . rpe_k^T ; tq = scale * k . rpe_q^T
    k_gemm_pos<<<BHN * 64 * 4, 32, 0, stream>>>(qB, rpekbf, tk, 1.0f);
    k_gemm_pos<<<BHN * 64 * 4, 32, 0, stream>>>(kB, rpeqbf, tq, scale);

    // fused attention: 64 bh x 64 row-blocks
    k_attn<<<BHN * 64, 32, 0, stream>>>(qB, kB, vT, tk, tq, rvt, ctx);

    // proj + residual + mean pool
    k_proj<<<4 * 512, 32, 0, stream>>>(ctx, projwbf, proj_b, x, pooled);

    // classification head
    k_head<<<1, 128, 0, stream>>>(pooled, fc_w, fc_b, out);
}
